// MultiHeadAttention_52407190946074
// MI455X (gfx1250) — compile-verified, matrix operand order checked by reading, not run on hardware
//
#include <hip/hip_runtime.h>

#define B_ 2
#define S_ 2048
#define E_ 1024
#define H_ 16
#define D_ 64

typedef __attribute__((ext_vector_type(16))) __bf16 v16bf;
typedef __attribute__((ext_vector_type(8)))  float  v8f;

__device__ __forceinline__ v8f wmma_bf16(v16bf a, v16bf b, v8f c) {
  // D = A(16x32 bf16) x B(32x16 bf16) + C(16x16 f32)
  return __builtin_amdgcn_wmma_f32_16x16x32_bf16(false, a, false, b, (short)0, c,
                                                 false, false);
}

// Load one wave32 A/B fragment row: 16 bf16 at k = {kb..kb+7, 16+kb..16+kb+7}
// p must already point at &tile[row_or_col][kb] (row stride handled by caller).
__device__ __forceinline__ v16bf ld_frag(const __bf16* p) {
  v16bf r;
#pragma unroll
  for (int j = 0; j < 8; ++j) { r[j] = p[j]; r[j + 8] = p[j + 16]; }
  return r;
}

// CDNA5 async global->LDS copy: 16B per lane, tracked by ASYNCcnt (not DScnt),
// so it is NOT waited by barriers -> true copy/compute overlap.
__device__ __forceinline__ void async_ld_b128(uint32_t lds_off, uint32_t goff,
                                              uint64_t sbase) {
  asm volatile("global_load_async_to_lds_b128 %0, %1, %2"
               :: "v"(lds_off), "v"(goff), "s"(sbase) : "memory");
}
__device__ __forceinline__ uint32_t lds_addr_of(const void* p) {
  // flat shared address = aperture | allocation-relative LDS offset (addr[31:0])
  return (uint32_t)(uintptr_t)p;
}

// ---------------------------------------------------------------------------
// Kernel 1: per-head projection  Y[b,h,s,:] = (X[b,s,h*64:...] * scale) @ W^T
// ---------------------------------------------------------------------------
__global__ __launch_bounds__(128) void qkv_project_kernel(
    const float* __restrict__ X, const float* __restrict__ W,
    __bf16* __restrict__ Y, float scale)
{
  __shared__ __bf16 sX[64][72];   // (m,k)
  __shared__ __bf16 sW[64][72];   // (n,k): B[k,n] = W[n][k]

  const int tid = threadIdx.x;
  const int st  = blockIdx.x & 31;
  const int bh  = blockIdx.x >> 5;
  const int h   = bh & (H_ - 1);
  const int b   = bh >> 4;
  const int s0  = st * 64;

  const float* Xb = X + ((size_t)(b * S_ + s0)) * E_ + h * D_;
  for (int i = tid; i < 64 * 64; i += 128) {
    int r = i >> 6, c = i & 63;
    sX[r][c] = (__bf16)(Xb[(size_t)r * E_ + c] * scale);
    sW[r][c] = (__bf16)W[i];
  }
  __syncthreads();

  const int lane = tid & 31;
  const int l16  = lane & 15;
  const int kb   = (lane >> 4) << 3;
  const int m0   = (tid >> 5) * 16;

  v8f acc[4] = {};
#pragma unroll
  for (int kk = 0; kk < 64; kk += 32) {
    v16bf a = ld_frag(&sX[m0 + l16][kk + kb]);
#pragma unroll
    for (int nt = 0; nt < 4; ++nt) {
      v16bf bf = ld_frag(&sW[nt * 16 + l16][kk + kb]);
      acc[nt] = wmma_bf16(a, bf, acc[nt]);
    }
  }

  __bf16* Yb = Y + ((size_t)bh * S_ + s0) * D_;
  const int mo = (lane >> 4) << 3;
#pragma unroll
  for (int nt = 0; nt < 4; ++nt)
#pragma unroll
    for (int r = 0; r < 8; ++r)
      Yb[(size_t)(m0 + r + mo) * D_ + nt * 16 + l16] = (__bf16)acc[nt][r];
}

// ---------------------------------------------------------------------------
// Kernel 2: flash attention. 256 thr (8 waves), 128 q-rows/block, 64-key tiles.
// Q pre-scaled by log2(e)/sqrt(D) -> softmax in exp2 domain.
// K tiles double-buffered via async global->LDS; V pipelined through VGPRs.
// ---------------------------------------------------------------------------
__global__ __launch_bounds__(256) void flash_attn_kernel(
    const __bf16* __restrict__ Qp, const __bf16* __restrict__ Kp,
    const __bf16* __restrict__ Vp, __bf16* __restrict__ Og)
{
  __shared__ __bf16 sQ[128][72];        // (m,d)
  __shared__ __bf16 sK[2][64][72];      // (key,d): B frag for scores
  __shared__ __bf16 sVt[2][64][72];     // (d,key): B frag for P@V
  __shared__ __bf16 sP[8][16][72];      // wave-private P (m,key)

  const int tid = threadIdx.x;
  const int st  = blockIdx.x & 15;      // S/128 = 16 q-tiles
  const int bh  = blockIdx.x >> 4;
  const int h   = bh & (H_ - 1);
  const int b   = bh >> 4;
  const int s0  = st * 128;

  const __bf16* Qb = Qp + ((size_t)bh * S_ + s0) * D_;
  for (int i = tid; i < 128 * 64; i += 256) sQ[i >> 6][i & 63] = Qb[i];

  const int lane = tid & 31;
  const int l16  = lane & 15;
  const int kb   = (lane >> 4) << 3;
  const int wave = tid >> 5;
  const int m0   = wave * 16;
  const int mo   = (lane >> 4) << 3;

  const __bf16* Kb = Kp + (size_t)bh * S_ * D_;
  const __bf16* Vb = Vp + (size_t)bh * S_ * D_;

  // ---- stage tile 0 -------------------------------------------------------
  const int c0 = tid, c1 = tid + 256;          // two 16B chunks per thread
  {
    uint32_t lk = lds_addr_of(&sK[0][0][0]);
    uint64_t gk = (uint64_t)(uintptr_t)Kb;
    async_ld_b128(lk + (c0 >> 3) * 144 + (c0 & 7) * 16, c0 * 16, gk);
    async_ld_b128(lk + (c1 >> 3) * 144 + (c1 & 7) * 16, c1 * 16, gk);
    const uint4* Vt = (const uint4*)Vb;
    uint4 q0 = Vt[c0], q1 = Vt[c1];
    const __bf16* p0 = (const __bf16*)&q0;
    const __bf16* p1 = (const __bf16*)&q1;
#pragma unroll
    for (int j = 0; j < 8; ++j) {
      sVt[0][(c0 & 7) * 8 + j][c0 >> 3] = p0[j];
      sVt[0][(c1 & 7) * 8 + j][c1 >> 3] = p1[j];
    }
  }
  __syncthreads();                             // sQ + V0 visible

  v16bf aQ0 = ld_frag(&sQ[m0 + l16][kb]);
  v16bf aQ1 = ld_frag(&sQ[m0 + l16][32 + kb]);

  v8f o[4] = {};
  float mrow[8], lrow[8];
#pragma unroll
  for (int r = 0; r < 8; ++r) { mrow[r] = -1e30f; lrow[r] = 0.0f; }

  const int NT = S_ / 64;                      // 32 key tiles
  for (int t = 0; t < NT; ++t) {
    const int buf = t & 1;
    uint4 q0, q1;
    if (t + 1 < NT) {                          // stage K[t+1] (async) + V[t+1] regs
      uint32_t lk = lds_addr_of(&sK[buf ^ 1][0][0]);
      uint64_t gk = (uint64_t)(uintptr_t)(Kb + (size_t)(t + 1) * 4096);
      async_ld_b128(lk + (c0 >> 3) * 144 + (c0 & 7) * 16, c0 * 16, gk);
      async_ld_b128(lk + (c1 >> 3) * 144 + (c1 & 7) * 16, c1 * 16, gk);
      const uint4* Vt = (const uint4*)(Vb + (size_t)(t + 1) * 4096);
      q0 = Vt[c0]; q1 = Vt[c1];
      asm volatile("s_wait_asynccnt 2" ::: "memory");   // own K[t] complete
    } else {
      asm volatile("s_wait_asynccnt 0" ::: "memory");
    }
    __syncthreads();                           // all waves: K[t]+V[t] in LDS

    // ---- scores: 16x64 per wave (4 tiles), d-reduction via 2 WMMAs each ---
    v8f tt[4] = {};
#pragma unroll
    for (int nt = 0; nt < 4; ++nt) {
      v16bf b0 = ld_frag(&sK[buf][nt * 16 + l16][kb]);
      v16bf b1 = ld_frag(&sK[buf][nt * 16 + l16][32 + kb]);
      tt[nt] = wmma_bf16(aQ0, b0, tt[nt]);
      tt[nt] = wmma_bf16(aQ1, b1, tt[nt]);
    }

    // ---- online softmax (exp2 domain); rows live in 16-lane halves --------
#pragma unroll
    for (int r = 0; r < 8; ++r) {
      float v = fmaxf(fmaxf(tt[0][r], tt[1][r]), fmaxf(tt[2][r], tt[3][r]));
      v = fmaxf(v, __shfl_xor(v, 1, 16));
      v = fmaxf(v, __shfl_xor(v, 2, 16));
      v = fmaxf(v, __shfl_xor(v, 4, 16));
      v = fmaxf(v, __shfl_xor(v, 8, 16));
      float mn   = fmaxf(mrow[r], v);
      float corr = exp2f(mrow[r] - mn);
      mrow[r] = mn;
      float p0 = exp2f(tt[0][r] - mn);
      float p1 = exp2f(tt[1][r] - mn);
      float p2 = exp2f(tt[2][r] - mn);
      float p3 = exp2f(tt[3][r] - mn);
      int m = r + mo;
      sP[wave][m][l16]      = (__bf16)p0;      // D-layout -> A-layout via LDS
      sP[wave][m][16 + l16] = (__bf16)p1;
      sP[wave][m][32 + l16] = (__bf16)p2;
      sP[wave][m][48 + l16] = (__bf16)p3;
      float rs = (p0 + p1) + (p2 + p3);
      rs += __shfl_xor(rs, 1, 16);
      rs += __shfl_xor(rs, 2, 16);
      rs += __shfl_xor(rs, 4, 16);
      rs += __shfl_xor(rs, 8, 16);
      lrow[r] = lrow[r] * corr + rs;
#pragma unroll
      for (int nt = 0; nt < 4; ++nt) o[nt][r] *= corr;
    }

    // ---- O += P(16x64) @ V(64x64); wave-local LDS is in-order -------------
    v16bf ap0 = ld_frag(&sP[wave][l16][kb]);
    v16bf ap1 = ld_frag(&sP[wave][l16][32 + kb]);
#pragma unroll
    for (int nt = 0; nt < 4; ++nt) {
      v16bf bv0 = ld_frag(&sVt[buf][nt * 16 + l16][kb]);
      v16bf bv1 = ld_frag(&sVt[buf][nt * 16 + l16][32 + kb]);
      o[nt] = wmma_bf16(ap0, bv0, o[nt]);
      o[nt] = wmma_bf16(ap1, bv1, o[nt]);
    }

    if (t + 1 < NT) {                          // commit V[t+1] regs -> LDS
      const __bf16* p0 = (const __bf16*)&q0;
      const __bf16* p1 = (const __bf16*)&q1;
#pragma unroll
      for (int j = 0; j < 8; ++j) {
        sVt[buf ^ 1][(c0 & 7) * 8 + j][c0 >> 3] = p0[j];
        sVt[buf ^ 1][(c1 & 7) * 8 + j][c1 >> 3] = p1[j];
      }
    }
    __syncthreads();                           // buffer-reuse fence (waits DS)
  }

  __bf16* Ob = Og + ((size_t)(b * S_ + s0)) * E_ + h * D_;
#pragma unroll
  for (int nt = 0; nt < 4; ++nt)
#pragma unroll
    for (int r = 0; r < 8; ++r)
      Ob[(size_t)(m0 + r + mo) * E_ + nt * 16 + l16] =
          (__bf16)(o[nt][r] / lrow[r]);
}

// ---------------------------------------------------------------------------
// Kernel 3: out = A(4096x1024 bf16) @ Wo^T + bo, fp32 result.
// ---------------------------------------------------------------------------
__global__ __launch_bounds__(128) void out_project_kernel(
    const __bf16* __restrict__ A, const float* __restrict__ Wo,
    const float* __restrict__ bo, float* __restrict__ out)
{
  __shared__ __bf16 sA[64][72];
  __shared__ __bf16 sW[64][72];

  const int tid   = threadIdx.x;
  const int ntile = blockIdx.x & 15;
  const int mtile = blockIdx.x >> 4;
  const int n0  = ntile * 64;
  const int m0g = mtile * 64;

  const int lane = tid & 31;
  const int l16  = lane & 15;
  const int kb   = (lane >> 4) << 3;
  const int m0   = (tid >> 5) * 16;

  v8f acc[4] = {};
  for (int kc = 0; kc < E_; kc += 64) {
    __syncthreads();
    for (int i = tid; i < 64 * 64; i += 128) {
      int r = i >> 6, c = i & 63;
      sA[r][c] = A[(size_t)(m0g + r) * E_ + kc + c];
      sW[r][c] = (__bf16)Wo[(size_t)(n0 + r) * E_ + kc + c];
    }
    __syncthreads();
#pragma unroll
    for (int kk = 0; kk < 64; kk += 32) {
      v16bf a = ld_frag(&sA[m0 + l16][kk + kb]);
#pragma unroll
      for (int nt = 0; nt < 4; ++nt) {
        v16bf bf = ld_frag(&sW[nt * 16 + l16][kk + kb]);
        acc[nt] = wmma_bf16(a, bf, acc[nt]);
      }
    }
  }

  const int mo = (lane >> 4) << 3;
#pragma unroll
  for (int nt = 0; nt < 4; ++nt)
#pragma unroll
    for (int r = 0; r < 8; ++r) {
      int m = m0g + m0 + r + mo;
      int n = n0 + nt * 16 + l16;
      out[(size_t)m * E_ + n] = acc[nt][r] + bo[n];
    }
}

extern "C" void kernel_launch(void* const* d_in, const int* in_sizes, int n_in,
                              void* d_out, int out_size, void* d_ws, size_t ws_size,
                              hipStream_t stream) {
  const float* values  = (const float*)d_in[0];
  const float* keys    = (const float*)d_in[1];
  const float* queries = (const float*)d_in[2];
  const float* Wv = (const float*)d_in[3];
  const float* Wk = (const float*)d_in[4];
  const float* Wq = (const float*)d_in[5];
  const float* Wo = (const float*)d_in[6];
  const float* bo = (const float*)d_in[7];
  float* out = (float*)d_out;

  const size_t elems = (size_t)B_ * S_ * E_;
  __bf16* Qp = (__bf16*)d_ws;
  __bf16* Kp = Qp + elems;
  __bf16* Vp = Kp + elems;
  __bf16* Og = Vp + elems;

  dim3 blk(128);
  dim3 grd(B_ * H_ * (S_ / 64));
  // Q scale: 1/sqrt(64) * log2(e)  (softmax computed in exp2 domain)
  hipLaunchKernelGGL(qkv_project_kernel, grd, blk, 0, stream, queries, Wq, Qp,
                     0.125f * 1.44269504088896340736f);
  hipLaunchKernelGGL(qkv_project_kernel, grd, blk, 0, stream, keys,   Wk, Kp, 1.0f);
  hipLaunchKernelGGL(qkv_project_kernel, grd, blk, 0, stream, values, Wv, Vp, 1.0f);
  hipLaunchKernelGGL(flash_attn_kernel, dim3(B_ * H_ * (S_ / 128)), dim3(256),
                     0, stream, Qp, Kp, Vp, Og);
  hipLaunchKernelGGL(out_project_kernel, dim3((B_ * S_ / 64) * (E_ / 64)), blk,
                     0, stream, Og, Wo, bo, out);
}
